// GemmaAttention_59648505807300
// MI455X (gfx1250) — matrix operand order repaired and checked by reading, not compiled, not run on hardware
//
#include <hip/hip_runtime.h>
#include <hip/hip_bf16.h>
#include <math.h>
#include <stdint.h>

typedef __bf16 bf16_t;
typedef __attribute__((ext_vector_type(16))) __bf16 v16bf;
typedef __attribute__((ext_vector_type(8)))  float  v8f;
typedef __attribute__((ext_vector_type(4)))  int    v4i;

union AFrag { v16bf v; uint4 u[2]; };

__device__ __forceinline__ v8f wmma_bf16(v16bf a, v16bf b, v8f c) {
  // (neg_a, A, neg_b, B, c_mod, C, reuse_a, reuse_b)
  return __builtin_amdgcn_wmma_f32_16x16x32_bf16(false, a, false, b, (short)0, c,
                                                 false, false);
}

// ---------------------------------------------------------------------------
// Async memory->LDS (GLOBAL_LOAD_ASYNC_TO_LDS_B128, ASYNCcnt-tracked)
// ---------------------------------------------------------------------------
typedef __attribute__((address_space(1))) v4i g_v4i;   // global (device) int4
typedef __attribute__((address_space(3))) v4i l_v4i;   // LDS (shared) int4

__device__ __forceinline__ void async_ld_b128(const bf16_t* g, bf16_t* l) {
#if __has_builtin(__builtin_amdgcn_global_load_async_to_lds_b128)
  __builtin_amdgcn_global_load_async_to_lds_b128(
      (g_v4i*)(uintptr_t)g, (l_v4i*)(uintptr_t)l, 0, 0);
#else
  asm volatile("global_load_async_to_lds_b128 %0, %1, off"
               :: "v"((unsigned)(uintptr_t)l), "v"(g)
               : "memory");
#endif
}

#if __has_builtin(__builtin_amdgcn_s_wait_asynccnt)
#define WAIT_ASYNC(n) __builtin_amdgcn_s_wait_asynccnt(n)
#else
#define WAIT_ASYNC(n) asm volatile("s_wait_asynccnt " #n ::: "memory")
#endif

// ---------------------------------------------------------------------------
// Stage 0: fp32 -> bf16 conversions (plain and transposing for weights)
// ---------------------------------------------------------------------------
__global__ void cvt_kernel(const float* __restrict__ src, bf16_t* __restrict__ dst, int n) {
  int i = blockIdx.x * 256 + threadIdx.x;
  if (i < n) dst[i] = (bf16_t)src[i];
}

// src is (K x N) row-major fp32; dst is (N x K) row-major bf16
__global__ void cvtT_kernel(const float* __restrict__ src, bf16_t* __restrict__ dst,
                            int K, int N) {
  int i = blockIdx.x * 256 + threadIdx.x;
  if (i >= K * N) return;
  int k = i / N, n = i % N;
  dst[(size_t)n * K + k] = (bf16_t)src[i];
}

// ---------------------------------------------------------------------------
// Stage 1/4: GEMM  C[MxN] = A[MxK](bf16) * Bt[NxK]^T(bf16), f32 accumulate.
// 128x128 block tile, 256 threads = 8 waves, each wave 64(M)x32(N).
// Double-buffered LDS tiles filled with async-to-LDS copies.
// ---------------------------------------------------------------------------
template <bool F32OUT>
__global__ __launch_bounds__(256)
void gemm_bt_kernel(const bf16_t* __restrict__ A, const bf16_t* __restrict__ Bt,
                    void* __restrict__ Cout, int M, int N, int K) {
  __shared__ bf16_t As[2][128][40];   // 32 K-cols + 8 pad (bank-conflict avoidance)
  __shared__ bf16_t Bs[2][128][40];
  const int t = threadIdx.x;
  const int lane = t & 31, w = t >> 5;
  const int wm = w & 1, wn = w >> 1;          // 2x4 wave grid
  const int lrow = lane & 15, lhi = lane >> 4;
  const int m0 = blockIdx.y * 128;
  const int n0 = blockIdx.x * 128;

  v8f acc[4][2];
#pragma unroll
  for (int mf = 0; mf < 4; ++mf)
#pragma unroll
    for (int nf = 0; nf < 2; ++nf) acc[mf][nf] = {};

  // each thread owns one 32B segment of one row of each tile
  const int row = t >> 1, seg = (t & 1) * 16;
  const bf16_t* ga = A  + (size_t)(m0 + row) * K + seg;
  const bf16_t* gb = Bt + (size_t)(n0 + row) * K + seg;

#define GEMM_ISSUE(K0, BUF)                                   \
  do {                                                        \
    async_ld_b128(ga + (K0),     &As[BUF][row][seg]);         \
    async_ld_b128(ga + (K0) + 8, &As[BUF][row][seg + 8]);     \
    async_ld_b128(gb + (K0),     &Bs[BUF][row][seg]);         \
    async_ld_b128(gb + (K0) + 8, &Bs[BUF][row][seg + 8]);     \
  } while (0)

  const int nIters = K >> 5;
  GEMM_ISSUE(0, 0);

  for (int it = 0; it < nIters; ++it) {
    const int cur = it & 1;
    const bool hasNext = (it + 1) < nIters;
    if (hasNext) {
      GEMM_ISSUE((it + 1) * 32, cur ^ 1);   // prefetch next tile into other buffer
      WAIT_ASYNC(4);                        // current tile's 4 ops complete (in-order)
    } else {
      WAIT_ASYNC(0);
    }
    __syncthreads();

    AFrag a[4], b[2];
#pragma unroll
    for (int mf = 0; mf < 4; ++mf) {
      const bf16_t* p = &As[cur][wm * 64 + mf * 16 + lrow][lhi * 8];
      a[mf].u[0] = *(const uint4*)p;
      a[mf].u[1] = *(const uint4*)(p + 16);
    }
#pragma unroll
    for (int nf = 0; nf < 2; ++nf) {
      const bf16_t* p = &Bs[cur][wn * 32 + nf * 16 + lrow][lhi * 8];
      b[nf].u[0] = *(const uint4*)p;
      b[nf].u[1] = *(const uint4*)(p + 16);
    }
#pragma unroll
    for (int mf = 0; mf < 4; ++mf)
#pragma unroll
      for (int nf = 0; nf < 2; ++nf)
        acc[mf][nf] = wmma_bf16(a[mf].v, b[nf].v, acc[mf][nf]);
    __syncthreads();   // all waves done reading buf `cur` before it is refilled
  }
#undef GEMM_ISSUE

#pragma unroll
  for (int mf = 0; mf < 4; ++mf)
#pragma unroll
    for (int nf = 0; nf < 2; ++nf)
#pragma unroll
      for (int r = 0; r < 8; ++r) {
        int m = m0 + wm * 64 + mf * 16 + r + lhi * 8;
        int n = n0 + wn * 32 + nf * 16 + lrow;
        float v = acc[mf][nf][r];
        if (F32OUT) ((float*)Cout)[(size_t)m * N + n] = v;
        else        ((bf16_t*)Cout)[(size_t)m * N + n] = (bf16_t)v;
      }
}

// ---------------------------------------------------------------------------
// Stage 2: RoPE in-place on qkv (cols 0..2047 = 8 q heads, 2048..2303 = k)
// ---------------------------------------------------------------------------
__global__ void rope_kernel(bf16_t* __restrict__ qkv, const int* __restrict__ pos) {
  int idx = blockIdx.x * 256 + threadIdx.x;       // 4096 rows * 9 heads * 128 pairs
  int i = idx & 127;
  int tt = idx >> 7;
  int head = tt % 9;
  int row = tt / 9;
  if (row >= 4096) return;
  int colbase = (head < 8) ? head * 256 : 2048;
  float p = (float)pos[row];
  float inv = __powf(10000.0f, -(float)i * (1.0f / 128.0f));
  float ang = p * inv;
  float c = __cosf(ang), s = __sinf(ang);
  size_t base = (size_t)row * 2560 + colbase;
  float x1 = (float)qkv[base + i];
  float x2 = (float)qkv[base + i + 128];
  qkv[base + i]       = (bf16_t)(x1 * c - x2 * s);
  qkv[base + i + 128] = (bf16_t)(x2 * c + x1 * s);
}

// ---------------------------------------------------------------------------
// Stage 3: flash attention. Grid (b*8+h, qtile). 256 thr = 8 waves x 16 q-rows.
// K tile staged with async-to-LDS; V tile transposed through registers.
// ---------------------------------------------------------------------------
__global__ __launch_bounds__(256)
void attn_kernel(const bf16_t* __restrict__ qkv, const float* __restrict__ mask,
                 bf16_t* __restrict__ attn) {
  __shared__ bf16_t Ks[32][264];      // 32 keys x 256 d (+pad)
  __shared__ bf16_t Vt[256][40];      // d x 32 keys (+pad), transposed
  __shared__ bf16_t Ps[8][16][32];    // per-wave P relayout buffer
  const int t = threadIdx.x, lane = t & 31, w = t >> 5;
  const int lrow = lane & 15, lhi = lane >> 4;
  const int b = blockIdx.x >> 3, h = blockIdx.x & 7;
  const int q0 = blockIdx.y * 128 + w * 16;
  const size_t RS = 2560;             // qkv row stride (elements)

  // Q fragments resident for the whole key loop: 16 rows x 256 d = 8 A-frags
  AFrag qf[8];
  {
    const bf16_t* qrowp = qkv + ((size_t)b * 2048 + q0 + lrow) * RS + h * 256;
#pragma unroll
    for (int kc = 0; kc < 8; ++kc) {
      qf[kc].u[0] = *(const uint4*)(qrowp + kc * 32 + lhi * 8);
      qf[kc].u[1] = *(const uint4*)(qrowp + kc * 32 + lhi * 8 + 16);
    }
  }

  v8f acc[16];
#pragma unroll
  for (int j = 0; j < 16; ++j) acc[j] = {};
  float mrun[8], lsum[8];
#pragma unroll
  for (int r = 0; r < 8; ++r) { mrun[r] = -__builtin_inff(); lsum[r] = 0.f; }

  for (int kb = 0; kb < 64; ++kb) {
    { // K tile: 32 keys x 256 d, row-major, async memory->LDS (4x b128/thread)
      int key = t >> 3, dseg = (t & 7) * 32;
      const bf16_t* src = qkv + ((size_t)b * 2048 + kb * 32 + key) * RS + 2048 + dseg;
      bf16_t* dst = &Ks[key][dseg];
      async_ld_b128(src,      dst);
      async_ld_b128(src + 8,  dst + 8);
      async_ld_b128(src + 16, dst + 16);
      async_ld_b128(src + 24, dst + 24);
    }
    { // V tile transposed: Vt[d][key] (contiguous B-frags for P*V)
      int key = t & 31, dbase = (t >> 5) * 32;
      const uint4* src =
          (const uint4*)(qkv + ((size_t)b * 2048 + kb * 32 + key) * RS + 2304 + dbase);
      union { uint4 u[4]; bf16_t e[32]; } vb;
      vb.u[0] = src[0]; vb.u[1] = src[1]; vb.u[2] = src[2]; vb.u[3] = src[3];
#pragma unroll
      for (int j = 0; j < 32; ++j) Vt[dbase + j][key] = vb.e[j];
    }
    WAIT_ASYNC(0);
    __syncthreads();

    // scores: S(16x32) = Q(16x256) x K^T, contract d in 8 WMMA steps
    v8f sc[2]; sc[0] = {}; sc[1] = {};
#pragma unroll
    for (int kc = 0; kc < 8; ++kc)
#pragma unroll
      for (int nf = 0; nf < 2; ++nf) {
        AFrag kf;
        const bf16_t* p = &Ks[nf * 16 + lrow][kc * 32 + lhi * 8];
        kf.u[0] = *(const uint4*)p;
        kf.u[1] = *(const uint4*)(p + 16);
        sc[nf] = wmma_bf16(qf[kc].v, kf.v, sc[nf]);
      }

    // online softmax (half-wave row reductions match C-fragment layout)
    float alpha[8];
#pragma unroll
    for (int r = 0; r < 8; ++r) {
      int qrow = blockIdx.y * 128 + w * 16 + r + lhi * 8;
      const float* mrow = mask + ((size_t)b * 2048 + qrow) * 2048 + kb * 32;
      float s0 = sc[0][r] * 0.0625f + mrow[lrow];
      float s1 = sc[1][r] * 0.0625f + mrow[16 + lrow];
      float mx = fmaxf(s0, s1);
      mx = fmaxf(mx, __shfl_xor(mx, 1, 32));
      mx = fmaxf(mx, __shfl_xor(mx, 2, 32));
      mx = fmaxf(mx, __shfl_xor(mx, 4, 32));
      mx = fmaxf(mx, __shfl_xor(mx, 8, 32));
      float mn = fmaxf(mrun[r], mx);
      float al = __expf(mrun[r] - mn);
      float p0 = __expf(s0 - mn);
      float p1 = __expf(s1 - mn);
      float ps = p0 + p1;
      ps += __shfl_xor(ps, 1, 32);
      ps += __shfl_xor(ps, 2, 32);
      ps += __shfl_xor(ps, 4, 32);
      ps += __shfl_xor(ps, 8, 32);
      lsum[r] = lsum[r] * al + ps;
      mrun[r] = mn;
      alpha[r] = al;
      Ps[w][r + lhi * 8][lrow]      = (bf16_t)p0;
      Ps[w][r + lhi * 8][16 + lrow] = (bf16_t)p1;
    }
#pragma unroll
    for (int j = 0; j < 16; ++j)
#pragma unroll
      for (int r = 0; r < 8; ++r) acc[j][r] *= alpha[r];

    // P (C-layout) -> A-layout via per-wave LDS buffer
    AFrag pf;
    {
      const bf16_t* pp = &Ps[w][lrow][lhi * 8];
      pf.u[0] = *(const uint4*)pp;
      pf.u[1] = *(const uint4*)(pp + 16);
    }
    // O(16x256) += P(16x32) x V(32x256), 16 d-chunks
#pragma unroll
    for (int j = 0; j < 16; ++j) {
      AFrag vf;
      const bf16_t* vp = &Vt[j * 16 + lrow][lhi * 8];
      vf.u[0] = *(const uint4*)vp;
      vf.u[1] = *(const uint4*)(vp + 16);
      acc[j] = wmma_bf16(pf.v, vf.v, acc[j]);
    }
    __syncthreads();
  }

  // normalize + store to attn[b][s][h*256+d] (bf16)
#pragma unroll
  for (int j = 0; j < 16; ++j)
#pragma unroll
    for (int r = 0; r < 8; ++r) {
      int qrow = blockIdx.y * 128 + w * 16 + r + lhi * 8;
      float o = acc[j][r] / lsum[r];
      attn[((size_t)b * 2048 + qrow) * 2048 + h * 256 + j * 16 + lrow] = (bf16_t)o;
    }
}

// ---------------------------------------------------------------------------
extern "C" void kernel_launch(void* const* d_in, const int* in_sizes, int n_in,
                              void* d_out, int out_size, void* d_ws, size_t ws_size,
                              hipStream_t stream) {
  const float* hs   = (const float*)d_in[0];   // (2,2048,2048)
  const float* mask = (const float*)d_in[1];   // (2,1,2048,2048)
  const int*   pos  = (const int*)d_in[2];     // (2,2048)
  const float* wq   = (const float*)d_in[3];   // (2048,2048)
  const float* wk   = (const float*)d_in[4];   // (2048,256)
  const float* wv   = (const float*)d_in[5];   // (2048,256)
  const float* wo   = (const float*)d_in[6];   // (2048,2048)
  float* out = (float*)d_out;                  // (2,2048,2048)

  const int M = 4096, H = 2048, NQKV = 2560;
  char* ws = (char*)d_ws;
  bf16_t* hsb    = (bf16_t*)(ws);                       // 4096x2048      16.8MB
  bf16_t* wqkvT  = (bf16_t*)(ws + 16777216);            // 2560x2048      10.5MB
  bf16_t* woT    = (bf16_t*)(ws + 27262976);            // 2048x2048       8.4MB
  bf16_t* qkv    = (bf16_t*)(ws + 35651584);            // 4096x2560      21.0MB
  bf16_t* attn   = (bf16_t*)(ws + 56623104);            // 4096x2048      16.8MB

  // stage 0: convert / transpose
  cvt_kernel<<<(M * H + 255) / 256, 256, 0, stream>>>(hs, hsb, M * H);
  cvtT_kernel<<<(H * 2048 + 255) / 256, 256, 0, stream>>>(wq, wqkvT, H, 2048);
  cvtT_kernel<<<(H * 256 + 255) / 256, 256, 0, stream>>>(wk, wqkvT + (size_t)2048 * H, H, 256);
  cvtT_kernel<<<(H * 256 + 255) / 256, 256, 0, stream>>>(wv, wqkvT + (size_t)2304 * H, H, 256);
  cvtT_kernel<<<(H * 2048 + 255) / 256, 256, 0, stream>>>(wo, woT, 2048, H);

  // stage 1: fused QKV projection (WMMA, async double-buffered)
  gemm_bt_kernel<false><<<dim3(NQKV / 128, M / 128), 256, 0, stream>>>(
      hsb, wqkvT, qkv, M, NQKV, H);

  // stage 2: RoPE on q + k
  rope_kernel<<<(4096 * 9 * 128) / 256, 256, 0, stream>>>(qkv, pos);

  // stage 3: flash attention (WMMA + async K staging)
  attn_kernel<<<dim3(16, 16), 256, 0, stream>>>(qkv, mask, attn);

  // stage 4: output projection (WMMA, f32 out)
  gemm_bt_kernel<true><<<dim3(H / 128, M / 128), 256, 0, stream>>>(
      attn, woT, out, M, H, H);
}
